// TwoChannelHankelTraceLIF_MinWTA_17059610100028
// MI455X (gfx1250) — matrix operand order repaired and checked by reading, not compiled, not run on hardware
//
#include <hip/hip_runtime.h>
#include <hip/hip_bf16.h>

typedef __attribute__((ext_vector_type(16))) _Float16 v16h;
typedef __attribute__((ext_vector_type(8)))  float    v8f;

// Problem constants (from reference)
constexpr int Bn = 32, Ln = 2048, Hn = 64, Kn = 64;
constexpr int LcN = Ln - Hn + 1;                       // 1985 cropped timesteps
constexpr size_t Nsz = (size_t)Bn * LcN * Kn;          // per-channel [B,Lc,K] size
constexpr size_t CH   = 6*Nsz + Bn + (size_t)Bn*Kn + 1;
constexpr size_t OFF_D2  = 0;
constexpr size_t OFF_Z   = Nsz;
constexpr size_t OFF_SP  = 2*Nsz;
constexpr size_t OFF_SER = 3*Nsz;
constexpr size_t OFF_V   = 4*Nsz;
constexpr size_t OFF_IDX = 5*Nsz;
constexpr size_t OFF_W   = 5*Nsz + Bn;
constexpr size_t OFF_G   = 5*Nsz + Bn + (size_t)Bn*Kn;
constexpr size_t OFF_WRM = 6*Nsz + Bn + (size_t)Bn*Kn;

constexpr float LAM = 0.95f, TAUc = 20.0f, THc = 1.5f, TAU_A = 100.0f,
                BETA_A = 1.5f, V_RESET = -0.5f, THR = 0.5f;

// ---------------------------------------------------------------------------
// Kernel 1: Hankel GEMM via WMMA.
// Block = 256 threads = 8 waves. Each wave computes a 16(t) x 64(k) tile:
//   D = A(16x32 f16 Hankel slice of x) x B(32x16 f16 = W^T) , 2 K-chunks,
//   4 N-tiles -> 8 v_wmma_f32_16x16x32_f16 per wave.
// x window and W are staged in LDS as f16 (Hankel reuse: 64x).
// ---------------------------------------------------------------------------
__global__ __launch_bounds__(256)
void hankel_wmma(const float* __restrict__ x0, const float* __restrict__ x1,
                 const float* __restrict__ W0, const float* __restrict__ W1,
                 const float* __restrict__ bb0, const float* __restrict__ bb1,
                 float* __restrict__ out)
{
    const int c  = blockIdx.z;
    const int b  = blockIdx.y;
    const int t0 = blockIdx.x * 128;

    const float* x    = c ? x1  : x0;
    const float* W    = c ? W1  : W0;
    const float* bias = c ? bb1 : bb0;
    float* d2 = out + (size_t)c * CH + OFF_D2;

    __shared__ _Float16 sW[Kn * Hn];   // W[k][i] as f16, row-major (8 KB)
    __shared__ _Float16 sx[192];       // x[b, t0 .. t0+190] as f16
    __shared__ float    sb[Kn];

    const int tid = threadIdx.x;
    for (int i = tid; i < Kn * Hn; i += 256) sW[i] = (_Float16)W[i];
    for (int j = tid; j < 192; j += 256) {
        int gt = t0 + j;
        sx[j] = (_Float16)((gt < Ln) ? x[(size_t)b * Ln + gt] : 0.0f);
    }
    if (tid < Kn) sb[tid] = bias[tid];
    __syncthreads();

    const int lane = tid & 31;
    const int wv   = tid >> 5;       // wave id: 8 waves * 16 t-rows each
    const int m    = lane & 15;      // A-matrix row within tile
    const int hi   = lane >> 4;      // half-wave selector
    const int nloc = lane & 15;      // B/C/D column within tile

    v8f acc[4] = {};                 // 4 N-tiles of 16x16 f32

#pragma unroll
    for (int g = 0; g < 2; ++g) {    // K-chunks of 32
        // A fragment: 16-bit A 16x32 layout (ISA 7.12.2)
        v16h af;
#pragma unroll
        for (int e = 0; e < 16; ++e) {
            int v  = e >> 1;
            int kl = ((v >> 2) << 4) + (hi << 3) + ((v & 3) << 1) + (e & 1);
            af[e] = sx[wv * 16 + m + g * 32 + kl];   // Hankel: A[m, kl]=x[t0+wv*16+m+kl]
        }
#pragma unroll
        for (int nt = 0; nt < 4; ++nt) {
            // B fragment: lane holds col n = nloc, K = 8*hi+e (e<8), 16+8*hi+(e-8) (e>=8)
            v16h bf;
            const _Float16* wp = &sW[(nt * 16 + nloc) * Hn + g * 32 + hi * 8];
#pragma unroll
            for (int e = 0; e < 16; ++e) bf[e] = wp[e < 8 ? e : e + 8];
            acc[nt] = __builtin_amdgcn_wmma_f32_16x16x32_f16(
                false, af, false, bf, (short)0, acc[nt], false, false);
        }
    }

    // Epilogue: bias + ReLU, store D (32-bit C/D 16x16 layout)
#pragma unroll
    for (int nt = 0; nt < 4; ++nt) {
        const int k  = nt * 16 + nloc;
        const float bv = sb[k];
#pragma unroll
        for (int r = 0; r < 8; ++r) {
            const int t = t0 + wv * 16 + r + hi * 8;
            if (t < LcN) {
                float val = acc[nt][r] + bv;
                d2[((size_t)b * LcN + t) * Kn + k] = fmaxf(val, 0.0f);
            }
        }
    }
}

// ---------------------------------------------------------------------------
// Kernel 2: fused EMA trace + adaptive LIF, sequential over t, one thread per
// (channel, b, k). Streams d2 in, streams z / sp / ser / v out (non-temporal),
// and records t_first + total spike count for the WTA stage.
// ---------------------------------------------------------------------------
__global__ __launch_bounds__(256)
void scan_lif(float* __restrict__ out, int* __restrict__ tf_ws,
              float* __restrict__ sn_ws)
{
    const int gid = blockIdx.x * 256 + threadIdx.x;   // 0..4095
    const int c = gid >> 11;
    const int r = gid & 2047;
    const int b = r >> 6;
    const int k = r & 63;

    const size_t chan = (size_t)c * CH;
    const size_t base = ((size_t)b * LcN) * Kn + k;
    const float* d2 = out + chan + OFF_D2  + base;
    float* zp = out + chan + OFF_Z   + base;
    float* sp = out + chan + OFF_SP  + base;
    float* sr = out + chan + OFF_SER + base;
    float* vp = out + chan + OFF_V   + base;

    float z = 0.0f, v = 0.0f, a = 0.0f, snum = 0.0f;
    int tfirst = LcN;
    for (int t = 0; t < LcN; ++t) {
        const size_t o = (size_t)t * Kn;
        __builtin_prefetch(d2 + o + 16 * Kn, 0, 0);   // global_prefetch_b8
        const float d = d2[o];
        z = LAM * z + (1.0f - LAM) * d;               // per-filter trace
        __builtin_nontemporal_store(z, zp + o);
        const float th_eff = THc + BETA_A * a;
        v = v - v * (1.0f / TAUc) + z;                // membrane before reset
        const float s = (v >= th_eff) ? 1.0f : 0.0f;  // STE forward value
        snum += s;
        __builtin_nontemporal_store(s,    sp + o);
        __builtin_nontemporal_store(snum, sr + o);
        __builtin_nontemporal_store(v,    vp + o);
        v = v * (1.0f - s) + V_RESET * s;             // detached reset
        a = a - a * (1.0f / TAU_A) + s;
        tfirst = (s > THR && tfirst == LcN) ? t : tfirst;
    }
    tf_ws[gid] = tfirst;
    sn_ws[gid] = snum;
}

// ---------------------------------------------------------------------------
// Kernel 3: WTA reduction per (channel, b). Forward values of w collapse to
// the hard one-hot; idx = first-spiking filter (earliest t, smallest k) or
// argmax spike count fallback.
// ---------------------------------------------------------------------------
__global__ __launch_bounds__(64)
void wta_reduce(float* __restrict__ out, const int* __restrict__ tf_ws,
                const float* __restrict__ sn_ws, int* __restrict__ idx_ws)
{
    const int cb = blockIdx.x;           // c*32 + b
    const int c = cb >> 5, b = cb & 31;
    const int k = threadIdx.x;

    __shared__ int   stf[64];
    __shared__ float ssn[64];
    __shared__ int   sidx;

    const int g = cb * 64 + k;           // matches scan_lif gid layout
    stf[k] = tf_ws[g];
    ssn[k] = sn_ws[g];
    __syncthreads();

    if (k == 0) {
        int tstar = LcN;
        for (int i = 0; i < 64; ++i) tstar = (stf[i] < tstar) ? stf[i] : tstar;
        int kstar = 0;
        for (int i = 63; i >= 0; --i) if (stf[i] == tstar) kstar = i;
        float bestv = -1.0f; int kfb = 0;
        for (int i = 0; i < 64; ++i)
            if (ssn[i] > bestv) { bestv = ssn[i]; kfb = i; }
        const int idx = (tstar < LcN) ? kstar : kfb;
        sidx = idx;
        idx_ws[cb] = idx;
        out[(size_t)c * CH + OFF_IDX + b] = (float)idx;
        if (b == 0) out[(size_t)c * CH + OFF_WRM] = (float)(Hn - 1);
    }
    __syncthreads();
    out[(size_t)c * CH + OFF_W + (size_t)b * Kn + k] = (k == sidx) ? 1.0f : 0.0f;
}

// ---------------------------------------------------------------------------
// Kernel 4: gated = spikes masked to the winning column (forward value of
// spikes * w_hard). Pure streaming pass.
// ---------------------------------------------------------------------------
__global__ __launch_bounds__(256)
void gated_write(float* __restrict__ out, const int* __restrict__ idx_ws)
{
    const size_t gid = (size_t)blockIdx.x * 256 + threadIdx.x;
    if (gid >= 2 * Nsz) return;
    const int c = (int)(gid / Nsz);
    const size_t r = gid - (size_t)c * Nsz;      // (b*Lc + t)*K + k
    const int k = (int)(r & 63);
    const int b = (int)(r / ((size_t)LcN * Kn));
    const int idx = idx_ws[c * 32 + b];
    const size_t chan = (size_t)c * CH;
    const float s = __builtin_nontemporal_load(out + chan + OFF_SP + r);
    __builtin_nontemporal_store((k == idx) ? s : 0.0f, out + chan + OFF_G + r);
}

// ---------------------------------------------------------------------------
extern "C" void kernel_launch(void* const* d_in, const int* in_sizes, int n_in,
                              void* d_out, int out_size, void* d_ws, size_t ws_size,
                              hipStream_t stream)
{
    (void)in_sizes; (void)n_in; (void)out_size; (void)ws_size;
    const float* x  = (const float*)d_in[0];
    const float* y  = (const float*)d_in[1];
    const float* Wx = (const float*)d_in[2];
    const float* bx = (const float*)d_in[3];
    const float* Wy = (const float*)d_in[4];
    const float* by = (const float*)d_in[5];
    float* out = (float*)d_out;

    int*   tf_ws  = (int*)d_ws;
    float* sn_ws  = (float*)((char*)d_ws + 4096 * sizeof(int));
    int*   idx_ws = (int*)((char*)d_ws + 4096 * (sizeof(int) + sizeof(float)));

    // 1) Hankel conv as WMMA GEMM: grid = (t-chunks of 128, B, channels)
    dim3 g1((LcN + 127) / 128, Bn, 2);
    hipLaunchKernelGGL(hankel_wmma, g1, dim3(256), 0, stream,
                       x, y, Wx, Wy, bx, by, out);

    // 2) fused trace + LIF scan: 2*B*K = 4096 threads
    hipLaunchKernelGGL(scan_lif, dim3(16), dim3(256), 0, stream,
                       out, tf_ws, sn_ws);

    // 3) WTA per (channel, b)
    hipLaunchKernelGGL(wta_reduce, dim3(64), dim3(64), 0, stream,
                       out, tf_ws, sn_ws, idx_ws);

    // 4) gated output
    const size_t total = 2 * Nsz;
    hipLaunchKernelGGL(gated_write, dim3((unsigned)((total + 255) / 256)),
                       dim3(256), 0, stream, out, idx_ws);
}
